// NonLocalModule_46145128629039
// MI455X (gfx1250) — matrix operand order repaired and checked by reading, not compiled, not run on hardware
//
#include <hip/hip_runtime.h>

// ---------------------------------------------------------------------------
// Non-Local block (self-attention) for MI455X / gfx1250, wave32 + WMMA bf16,
// async global->LDS staging (ASYNCcnt) + TDM tensor_load_to_lds (TENSORcnt),
// double buffering, sched_group_barrier-pipelined DS->WMMA bursts.
//   B=4, C=512, E=128, H=W=64, N=4096
// ---------------------------------------------------------------------------

#define B_ 4
#define C_ 512
#define E_ 128
#define N_ 4096

typedef __attribute__((ext_vector_type(16))) __bf16 v16bf;
typedef __attribute__((ext_vector_type(8)))  __bf16 v8bf;
typedef __attribute__((ext_vector_type(8)))  float  v8f;
typedef __attribute__((ext_vector_type(4)))  unsigned u32x4;
typedef __attribute__((ext_vector_type(8)))  unsigned u32x8;

// Scheduling-group pipeline hints (no instructions emitted).
// Masks: 0x008=MFMA/WMMA, 0x020=VMEM read, 0x100=DS read, 0x200=DS write.
#if __has_builtin(__builtin_amdgcn_sched_group_barrier)
#define SCHED_GROUP(mask, size) __builtin_amdgcn_sched_group_barrier((mask), (size), 0)
#else
#define SCHED_GROUP(mask, size)
#endif

union AF { v16bf v; v8bf h[2]; };

// A-matrix 16x32 bf16 layout: lane holds row M=lane%16; contiguous 16 K values
// starting at K = 16*(lane/16)
static __device__ inline v16bf ldfragA(const __bf16* p) {
    AF f;
    f.h[0] = *(const v8bf*)(p);
    f.h[1] = *(const v8bf*)(p + 8);
    return f.v;
}

// B-matrix 32x16 bf16 layout: lane holds col N=lane%16; contiguous 16 K values
// starting at K = 16*(lane/16)  (this code's order, same as ldfragA)
static __device__ inline v16bf ldfragB(const __bf16* p) {
    AF f;
    f.h[0] = *(const v8bf*)(p);
    f.h[1] = *(const v8bf*)(p + 8);
    return f.v;
}

static __device__ inline v8f vzero() {
    v8f z = {0.f, 0.f, 0.f, 0.f, 0.f, 0.f, 0.f, 0.f};
    return z;
}

static __device__ inline v8f wmma_bf16(v16bf a, v16bf b, v8f c) {
    return __builtin_amdgcn_wmma_f32_16x16x32_bf16(
        false, a, false, b, (short)0, c, false, false);
}

// ---- CDNA5 async global->LDS copy (ASYNCcnt) ------------------------------
static __device__ inline void async_b128(const void* gptr, void* lptr) {
    asm volatile("global_load_async_to_lds_b128 %0, %1, off"
                 :: "v"((unsigned)(size_t)lptr), "v"(gptr)
                 : "memory");
}
static __device__ inline void wait_async0() {
    asm volatile("s_wait_asynccnt 0x0" ::: "memory");
}

// ---- CDNA5 Tensor Data Mover: 2D strided tile -> LDS (TENSORcnt) ----------
// Loads a [rows x 64B] tile whose rows are rowStrideBytes apart, as 8-byte
// elements. D# group0: count=1, lds_addr, 57-bit global_addr, type=2.
// D# group1: data_size=3 (8B), tensor_dim0=stride units, tensor_dim1=rows,
//            tile_dim0=8 (64B), tile_dim1=rows, tensor_dim0_stride in units.
static __device__ inline void tdm_load_rows64B(const void* gaddr, void* laddr,
                                               unsigned rows, unsigned rowStrideBytes) {
    const unsigned long long ga = (unsigned long long)gaddr;
    const unsigned sU = rowStrideBytes >> 3;     // stride in 8B units
    u32x4 g0;
    g0[0] = 1u;                                   // count=1, user mode
    g0[1] = (unsigned)(size_t)laddr;              // LDS byte address
    g0[2] = (unsigned)ga;                         // global_addr[31:0]
    g0[3] = (unsigned)(ga >> 32) | 0x80000000u;   // global_addr[56:32] | type=2
    u32x8 g1;
    g1[0] = 3u << 16;                             // data_size = 8 bytes
    g1[1] = (sU & 0xFFFFu) << 16;                 // tensor_dim0[15:0]
    g1[2] = (sU >> 16) | (rows << 16);            // tensor_dim0[31:16] | tensor_dim1[15:0]
    g1[3] = 8u << 16;                             // tensor_dim1[31:16]=0 | tile_dim0=8
    g1[4] = rows;                                 // tile_dim1=rows, tile_dim2=0
    g1[5] = sU;                                   // tensor_dim0_stride[31:0]
    g1[6] = 0u;
    g1[7] = 0u;
    asm volatile("tensor_load_to_lds %0, %1" :: "s"(g0), "s"(g1) : "memory");
}
static __device__ inline void wait_tensor0() {
    __builtin_amdgcn_s_wait_tensorcnt(0);
}

// 16-lane (half-wave) butterfly reductions via ds_swizzle.
#define SWZF(v, imm) __int_as_float(__builtin_amdgcn_ds_swizzle(__float_as_int(v), (imm)))

static __device__ inline float rowmax16(float v) {
    v = fmaxf(v, SWZF(v, 0x041F));
    v = fmaxf(v, SWZF(v, 0x081F));
    v = fmaxf(v, SWZF(v, 0x101F));
    v = fmaxf(v, SWZF(v, 0x201F));
    return v;
}
static __device__ inline float rowsum16(float v) {
    v += SWZF(v, 0x041F);
    v += SWZF(v, 0x081F);
    v += SWZF(v, 0x101F);
    v += SWZF(v, 0x201F);
    return v;
}

// ---------------------------------------------------------------------------
// Prep kernels
// ---------------------------------------------------------------------------

__global__ __launch_bounds__(256) void nl_transpose_x(const float* __restrict__ x,
                                                      __bf16* __restrict__ xT) {
    __shared__ float tile[32][33];
    const int tx = threadIdx.x & 31;
    const int ty = threadIdx.x >> 5;
    const int n0 = (blockIdx.x & (N_ / 32 - 1)) * 32;
    const int c0 = ((blockIdx.x / (N_ / 32)) & (C_ / 32 - 1)) * 32;
    const int b  = blockIdx.x / ((N_ / 32) * (C_ / 32));

#pragma unroll
    for (int i = 0; i < 4; ++i) {
        const int c = c0 + ty * 4 + i;
        tile[ty * 4 + i][tx] = x[((size_t)b * C_ + c) * N_ + n0 + tx];
    }
    __syncthreads();
#pragma unroll
    for (int i = 0; i < 4; ++i) {
        const int n = n0 + ty * 4 + i;
        xT[((size_t)b * N_ + n) * C_ + c0 + tx] = (__bf16)tile[tx][ty * 4 + i];
    }
}

__global__ void nl_cvt_bf16(const float* __restrict__ src, __bf16* __restrict__ dst, int n) {
    int i = blockIdx.x * blockDim.x + threadIdx.x;
    if (i < n) dst[i] = (__bf16)src[i];
}

__global__ void nl_bnfold(const float* __restrict__ bn, const float* __restrict__ bias,
                          float* __restrict__ sc, float* __restrict__ sh, int n) {
    int i = blockIdx.x * blockDim.x + threadIdx.x;
    if (i >= n) return;
    float gamma = bn[i], beta = bn[n + i], mean = bn[2 * n + i], var = bn[3 * n + i];
    float s = gamma * rsqrtf(var + 1e-5f);
    sc[i] = s;
    sh[i] = (bias[i] - mean) * s + beta;
}

// ---------------------------------------------------------------------------
// Kernel 1: fused theta/phi/g projections.
// ---------------------------------------------------------------------------
__global__ __launch_bounds__(128) void nl_qkv(
    const __bf16* __restrict__ xT,
    const __bf16* __restrict__ wq, const __bf16* __restrict__ wk,
    const __bf16* __restrict__ wv,
    const float* __restrict__ sc_q, const float* __restrict__ sh_q,
    const float* __restrict__ sc_k, const float* __restrict__ sh_k,
    const float* __restrict__ sc_v, const float* __restrict__ sh_v,
    __bf16* __restrict__ theta, __bf16* __restrict__ phi,
    __bf16* __restrict__ g) {
    __shared__ __bf16 swt[2][E_ * 32];

    const int tid  = threadIdx.x;
    const int wib  = tid >> 5;
    const int lane = tid & 31;
    const int lo = lane & 15, hi = lane >> 4;
    const int proj = blockIdx.x % 3;
    const int grp  = blockIdx.x / 3;
    const int b    = grp >> 6;
    const int n0   = (((grp & 63) << 2) + wib) << 4;

    const __bf16* w  = (proj == 0) ? wq : ((proj == 1) ? wk : wv);
    const float*  sc = (proj == 0) ? sc_q : ((proj == 1) ? sc_k : sc_v);
    const float*  sh = (proj == 0) ? sh_q : ((proj == 1) ? sh_k : sh_v);

    auto prefetch_w = [&](int kt, int bufi) {
        const char* gw = (const char*)(w + (size_t)tid * C_ + kt * 32);
        char* lw = (char*)(&swt[bufi][tid * 32]);
#pragma unroll
        for (int k = 0; k < 4; ++k) async_b128(gw + k * 16, lw + k * 16);
    };

    v8f acc[8];
#pragma unroll
    for (int et = 0; et < 8; ++et) acc[et] = vzero();

    const __bf16* xrow = xT + ((size_t)b * N_ + (n0 + lo)) * C_;

    prefetch_w(0, 0);
    int buf = 0;
    for (int kt = 0; kt < C_ / 32; ++kt) {
        wait_async0();
        __syncthreads();
        if (kt + 1 < C_ / 32) prefetch_w(kt + 1, buf ^ 1);

        v16bf a = ldfragA(xrow + kt * 32 + 16 * hi);
        const __bf16* sw = swt[buf];
#pragma unroll
        for (int eh = 0; eh < 2; ++eh) {
            v16bf bf[4];
#pragma unroll
            for (int e2 = 0; e2 < 4; ++e2)
                bf[e2] = ldfragB(sw + ((eh * 4 + e2) * 16 + lo) * 32 + 16 * hi);
#pragma unroll
            for (int e2 = 0; e2 < 4; ++e2)
                acc[eh * 4 + e2] = wmma_bf16(a, bf[e2], acc[eh * 4 + e2]);
        }
        // pipeline: A loads, then all 16 DS reads, then 8 WMMAs back-to-back
        SCHED_GROUP(0x020, 2);    // VMEM reads (A fragment)
        SCHED_GROUP(0x100, 16);   // DS reads (B fragments)
        SCHED_GROUP(0x008, 8);    // WMMA burst
        buf ^= 1;
    }

#pragma unroll
    for (int et = 0; et < 8; ++et) {
        const int e = et * 16 + lo;
        const float s = sc[e], t = sh[e];
#pragma unroll
        for (int j = 0; j < 8; ++j) {
            const int n = n0 + j + 8 * hi;
            const float v = acc[et][j] * s + t;
            if (proj == 2)
                g[((size_t)b * E_ + e) * N_ + n] = (__bf16)v;
            else if (proj == 0)
                theta[((size_t)b * N_ + n) * E_ + e] = (__bf16)v;
            else
                phi[((size_t)b * N_ + n) * E_ + e] = (__bf16)v;
        }
    }
}

// ---------------------------------------------------------------------------
// Kernel 2: flash attention. phi via per-lane async copies; g via TDM
// (wave 0). Double buffered; DS->WMMA bursts pinned by sched groups.
// ---------------------------------------------------------------------------
__global__ __launch_bounds__(128) void nl_attn(
    const __bf16* __restrict__ theta,   // [B][N][E]
    const __bf16* __restrict__ phi,     // [B][N][E]
    const __bf16* __restrict__ g,       // [B][E][N]
    __bf16* __restrict__ y) {           // [B][N][E]
    __shared__ __bf16 sphi[2][32 * E_];  // [m][e]
    __shared__ __bf16 sgt[2][E_ * 32];   // [e][m]
    __shared__ __bf16 pbuf[4][16 * 32];  // per-wave P staging

    const int tid  = threadIdx.x;
    const int wib  = tid >> 5;
    const int lane = tid & 31;
    const int lo = lane & 15, hi = lane >> 4;
    const int b  = blockIdx.x >> 6;
    const int n0 = (((blockIdx.x & 63) << 2) + wib) << 4;
    __bf16* pb = pbuf[wib];

    const __bf16* phibase = phi + (size_t)b * N_ * E_;
    const __bf16* gbase   = g + (size_t)b * E_ * N_;

    auto prefetch_kv = [&](int m0, int bufi) {
        const char* gp = (const char*)(phibase + (size_t)m0 * E_) + tid * 16;
        char* lp = (char*)(&sphi[bufi][0]) + tid * 16;
#pragma unroll
        for (int k = 0; k < 4; ++k) async_b128(gp + k * 2048, lp + k * 2048);
        if (wib == 0)
            tdm_load_rows64B((const char*)gbase + (size_t)m0 * 2, &sgt[bufi][0],
                             E_, N_ * 2);
    };

    v16bf at[4];
    {
        const __bf16* trow = theta + ((size_t)b * N_ + (n0 + lo)) * E_;
#pragma unroll
        for (int kt = 0; kt < 4; ++kt) at[kt] = ldfragA(trow + kt * 32 + 16 * hi);
    }

    v8f acc[8];
#pragma unroll
    for (int et = 0; et < 8; ++et) acc[et] = vzero();

    float m[8], l[8];
#pragma unroll
    for (int j = 0; j < 8; ++j) { m[j] = -1e30f; l[j] = 0.f; }

    prefetch_kv(0, 0);
    int buf = 0;
    for (int it = 0; it < N_ / 32; ++it) {
        wait_async0();
        if (wib == 0) wait_tensor0();
        __syncthreads();
        if (it + 1 < N_ / 32) prefetch_kv((it + 1) * 32, buf ^ 1);

        const __bf16* sp = sphi[buf];
        const __bf16* sg = sgt[buf];

        // ---- S = theta^T * phi : 16x32 logits ----
        v8f S0 = vzero(), S1 = vzero();
#pragma unroll
        for (int kh = 0; kh < 2; ++kh) {
            v16bf bp[4];
#pragma unroll
            for (int k2 = 0; k2 < 2; ++k2) {
                const int kt = kh * 2 + k2;
                bp[k2 * 2]     = ldfragB(sp + lo * E_ + kt * 32 + 16 * hi);
                bp[k2 * 2 + 1] = ldfragB(sp + (16 + lo) * E_ + kt * 32 + 16 * hi);
            }
#pragma unroll
            for (int k2 = 0; k2 < 2; ++k2) {
                const int kt = kh * 2 + k2;
                S0 = wmma_bf16(at[kt], bp[k2 * 2], S0);
                S1 = wmma_bf16(at[kt], bp[k2 * 2 + 1], S1);
            }
        }

        // ---- online softmax ----
        float corr[8];
#pragma unroll
        for (int j = 0; j < 8; ++j) {
            float rmax = rowmax16(fmaxf(S0[j], S1[j]));
            float nm = fmaxf(m[j], rmax);
            corr[j] = __expf(m[j] - nm);
            m[j] = nm;
        }
#pragma unroll
        for (int j = 0; j < 8; ++j) {
            float p0 = __expf(S0[j] - m[j]);
            float p1 = __expf(S1[j] - m[j]);
            l[j] = l[j] * corr[j] + rowsum16(p0 + p1);
            const int row = j + 8 * hi;
            pb[row * 32 + lo]      = (__bf16)p0;
            pb[row * 32 + 16 + lo] = (__bf16)p1;
        }
#pragma unroll
        for (int et = 0; et < 8; ++et) {
#pragma unroll
            for (int j = 0; j < 8; ++j) acc[et][j] *= corr[j];
        }

        // ---- acc += P x g^T ----
        v16bf ap = ldfragA(pb + lo * 32 + 16 * hi);
#pragma unroll
        for (int eh = 0; eh < 2; ++eh) {
            v16bf bg[4];
#pragma unroll
            for (int e2 = 0; e2 < 4; ++e2)
                bg[e2] = ldfragB(sg + ((eh * 4 + e2) * 16 + lo) * 32 + 16 * hi);
#pragma unroll
            for (int e2 = 0; e2 < 4; ++e2)
                acc[eh * 4 + e2] = wmma_bf16(ap, bg[e2], acc[eh * 4 + e2]);
        }

        // pipeline: S frag reads -> S WMMAs -> P stores -> PV reads -> PV WMMAs
        // (softmax VALU/trans ops left unconstrained to fill the gaps)
        SCHED_GROUP(0x100, 16);   // S-phase DS reads
        SCHED_GROUP(0x008, 8);    // S-phase WMMA burst
        SCHED_GROUP(0x200, 16);   // P staging DS writes
        SCHED_GROUP(0x100, 18);   // P A-frag + g B-frag DS reads
        SCHED_GROUP(0x008, 8);    // PV WMMA burst
        buf ^= 1;
    }

#pragma unroll
    for (int j = 0; j < 8; ++j) l[j] = 1.f / l[j];
#pragma unroll
    for (int et = 0; et < 8; ++et) {
        const int e = et * 16 + lo;
#pragma unroll
        for (int j = 0; j < 8; ++j) {
            const int n = n0 + j + 8 * hi;
            y[((size_t)b * N_ + n) * E_ + e] = (__bf16)(acc[et][j] * l[j]);
        }
    }
}

// ---------------------------------------------------------------------------
// Kernel 3: out = BN(w_w @ y + w_b) + x.
// ---------------------------------------------------------------------------
__global__ __launch_bounds__(128) void nl_proj(
    const __bf16* __restrict__ y,
    const __bf16* __restrict__ ww,
    const float* __restrict__ sc, const float* __restrict__ sh,
    const float* __restrict__ x,
    float* __restrict__ out) {
    __shared__ __bf16 sww[2][16 * E_];

    const int tid  = threadIdx.x;
    const int wib  = tid >> 5;
    const int lane = tid & 31;
    const int lo = lane & 15, hi = lane >> 4;
    const int b  = blockIdx.x >> 6;
    const int n0 = (((blockIdx.x & 63) << 2) + wib) << 4;

    auto prefetch_ww = [&](int ct, int bufi) {
#pragma unroll
        for (int k = 0; k < 2; ++k) {
            const int id = tid + k * 128;
            const int r = id >> 4, col = (id & 15) * 8;
            async_b128(ww + (size_t)(ct * 16 + r) * E_ + col, &sww[bufi][id * 8]);
        }
    };

    v16bf by[4];
    const __bf16* yrow = y + ((size_t)b * N_ + (n0 + lo)) * E_;
#pragma unroll
    for (int kt = 0; kt < 4; ++kt) by[kt] = ldfragB(yrow + kt * 32 + 16 * hi);

    prefetch_ww(0, 0);
    int buf = 0;
    for (int ct = 0; ct < C_ / 16; ++ct) {
        wait_async0();
        __syncthreads();
        if (ct + 1 < C_ / 16) prefetch_ww(ct + 1, buf ^ 1);

        v16bf aw[4];
#pragma unroll
        for (int kt = 0; kt < 4; ++kt)
            aw[kt] = ldfragA(sww[buf] + lo * E_ + kt * 32 + 16 * hi);
        v8f acc = vzero();
#pragma unroll
        for (int kt = 0; kt < 4; ++kt)
            acc = wmma_bf16(aw[kt], by[kt], acc);
#pragma unroll
        for (int j = 0; j < 8; ++j) {
            const int c = ct * 16 + j + 8 * hi;
            const int n = n0 + lo;
            const size_t idx = ((size_t)b * C_ + c) * (size_t)N_ + n;
            out[idx] = acc[j] * sc[c] + sh[c] + x[idx];
        }
        SCHED_GROUP(0x100, 8);    // DS reads (A fragments)
        SCHED_GROUP(0x008, 4);    // WMMA burst
        buf ^= 1;
    }
}

// ---------------------------------------------------------------------------
// Host launcher
// ---------------------------------------------------------------------------
extern "C" void kernel_launch(void* const* d_in, const int* in_sizes, int n_in,
                              void* d_out, int out_size, void* d_ws, size_t ws_size,
                              hipStream_t stream) {
    (void)in_sizes; (void)n_in; (void)out_size; (void)ws_size;

    const float* x        = (const float*)d_in[0];
    const float* theta_w  = (const float*)d_in[1];
    const float* theta_b  = (const float*)d_in[2];
    const float* theta_bn = (const float*)d_in[3];
    const float* phi_w    = (const float*)d_in[4];
    const float* phi_b    = (const float*)d_in[5];
    const float* phi_bn   = (const float*)d_in[6];
    const float* g_w      = (const float*)d_in[7];
    const float* g_b      = (const float*)d_in[8];
    const float* g_bn     = (const float*)d_in[9];
    const float* w_w      = (const float*)d_in[10];
    const float* w_b      = (const float*)d_in[11];
    const float* w_bn     = (const float*)d_in[12];
    float* out = (float*)d_out;

    char* ws = (char*)d_ws;
    size_t off = 0;
    auto alloc = [&](size_t bytes) -> void* {
        void* p = ws + off;
        off = (off + bytes + 255) & ~(size_t)255;
        return p;
    };

    __bf16* xT  = (__bf16*)alloc((size_t)B_ * N_ * C_ * 2);
    __bf16* th  = (__bf16*)alloc((size_t)B_ * N_ * E_ * 2);
    __bf16* ph  = (__bf16*)alloc((size_t)B_ * N_ * E_ * 2);
    __bf16* gg  = (__bf16*)alloc((size_t)B_ * E_ * N_ * 2);
    __bf16* yy  = (__bf16*)alloc((size_t)B_ * N_ * E_ * 2);
    __bf16* wq  = (__bf16*)alloc((size_t)E_ * C_ * 2);
    __bf16* wk  = (__bf16*)alloc((size_t)E_ * C_ * 2);
    __bf16* wv  = (__bf16*)alloc((size_t)E_ * C_ * 2);
    __bf16* wwb = (__bf16*)alloc((size_t)C_ * E_ * 2);
    float* sc_q = (float*)alloc(E_ * 4);
    float* sh_q = (float*)alloc(E_ * 4);
    float* sc_k = (float*)alloc(E_ * 4);
    float* sh_k = (float*)alloc(E_ * 4);
    float* sc_v = (float*)alloc(E_ * 4);
    float* sh_v = (float*)alloc(E_ * 4);
    float* sc_w = (float*)alloc(C_ * 4);
    float* sh_w = (float*)alloc(C_ * 4);

    nl_transpose_x<<<(N_ / 32) * (C_ / 32) * B_, 256, 0, stream>>>(x, xT);
    nl_cvt_bf16<<<(E_ * C_ + 255) / 256, 256, 0, stream>>>(theta_w, wq, E_ * C_);
    nl_cvt_bf16<<<(E_ * C_ + 255) / 256, 256, 0, stream>>>(phi_w,   wk, E_ * C_);
    nl_cvt_bf16<<<(E_ * C_ + 255) / 256, 256, 0, stream>>>(g_w,     wv, E_ * C_);
    nl_cvt_bf16<<<(C_ * E_ + 255) / 256, 256, 0, stream>>>(w_w,     wwb, C_ * E_);
    nl_bnfold<<<1, 128, 0, stream>>>(theta_bn, theta_b, sc_q, sh_q, E_);
    nl_bnfold<<<1, 128, 0, stream>>>(phi_bn,   phi_b,   sc_k, sh_k, E_);
    nl_bnfold<<<1, 128, 0, stream>>>(g_bn,     g_b,     sc_v, sh_v, E_);
    nl_bnfold<<<4, 128, 0, stream>>>(w_bn,     w_b,     sc_w, sh_w, C_);

    nl_qkv<<<768, 128, 0, stream>>>(xT, wq, wk, wv,
                                    sc_q, sh_q, sc_k, sh_k, sc_v, sh_v,
                                    th, ph, gg);
    nl_attn<<<256, 128, 0, stream>>>(th, ph, gg, yy);
    nl_proj<<<256, 128, 0, stream>>>(yy, wwb, sc_w, sh_w, x, out);
}
